// LegacyTemporalTransportGNN_18219251270345
// MI455X (gfx1250) — matrix operand order repaired and checked by reading, not compiled, not run on hardware
//
#include <hip/hip_runtime.h>
#include <math.h>

// ---------------------------------------------------------------------------
// CDNA5 / gfx1250 implementation of LegacyTemporalTransportGNN.
//   - all dense GEMMs via v_wmma_f32_16x16x32_bf16 (bf16 inputs, f32 accum)
//   - edge segment-sum via wave32 wave-per-edge float4 gather + f32 atomics
// ---------------------------------------------------------------------------

typedef __attribute__((ext_vector_type(16))) __bf16 v16bf;
typedef __attribute__((ext_vector_type(8)))  float  v8f;

#define NNODES 100000
#define HDIM   128
#define NEDGE  1600000
#define NB     65536
#define TDDIM  64

// ---------------------------------------------------------------------------
// Generic WMMA GEMM:  C[M x N] (+)= A[rowidx[m] or m][K] * W[K x N]
//   - NTILES (compile-time) 16-col output tiles per block: 8 (128-wide) or 4
//   - K multiple of 32, M multiple of 16
//   - W tile (32 x NTILES*16 cols) staged per k-step into LDS as bf16,
//     transposed to [n][k], padded row stride 36 to avoid bank conflicts
//   - zeroInit=1: acc starts at 0; zeroInit=0: acc initialized from C (fused
//     accumulation / bias-pre-initialized C)
// ---------------------------------------------------------------------------
template<int NTILES>
__global__ __launch_bounds__(256)
void gemm_bf16_wmma(const float* __restrict__ A, int lda,
                    const int* __restrict__ rowidx,
                    const float* __restrict__ W, int ldw,
                    float* __restrict__ C, int ldc,
                    int M, int K, int zeroInit)
{
    constexpr int NCOLS = NTILES * 16;
    __shared__ __bf16 Wt[NCOLS * 36];

    const int tid  = threadIdx.x;
    const int lane = tid & 31;
    const int wave = tid >> 5;          // wave32: 8 waves per 256-thread block
    const int hi   = lane >> 4;         // lane half (0/1)
    const int lo   = lane & 15;

    const int n0g = blockIdx.y * NCOLS;

    const int  m0     = blockIdx.x * 128 + wave * 16;
    const bool active = (m0 < M);       // wave-uniform -> EXEC all-1s at WMMA

    // A row handled by this lane (same row for both lane halves)
    const float* Arow = A;
    if (active) {
        int r = m0 + lo;
        int g = rowidx ? rowidx[r] : r;
        Arow = A + (size_t)g * lda;
    }

    v8f acc[NTILES] = {};
    if (active && !zeroInit) {
#pragma unroll
        for (int nt = 0; nt < NTILES; ++nt) {
            const float* Cp = C + (size_t)(m0 + hi * 8) * ldc + n0g + nt * 16 + lo;
#pragma unroll
            for (int r = 0; r < 8; ++r)
                acc[nt][r] = Cp[(size_t)r * ldc];
        }
    }

    const int kb = hi * 8;    // A-fragment K base within 32-wide k-tile
    const int kh = hi * 16;   // unused (B fragment also uses kb)
    const int nk = K >> 5;

    for (int kt = 0; kt < nk; ++kt) {
        __syncthreads();
        // stage W[kt*32 .. kt*32+31][n0g .. n0g+NCOLS) -> Wt[n][k] (bf16)
#pragma unroll
        for (int idx = tid; idx < (NCOLS << 5); idx += 256) {
            int n = idx >> 5, k = idx & 31;
            Wt[n * 36 + k] = (__bf16)W[(size_t)(kt * 32 + k) * ldw + n0g + n];
        }
        __syncthreads();
        if (!active) continue;

        // A fragment: elems 0..7 -> K = kb+0..7 ; elems 8..15 -> K = kb+16..23
        v16bf afrag;
        const float* Ap = Arow + kt * 32 + kb;
#pragma unroll
        for (int i = 0; i < 8; ++i) {
            afrag[i]     = (__bf16)Ap[i];
            afrag[i + 8] = (__bf16)Ap[i + 16];
        }

#pragma unroll
        for (int nt = 0; nt < NTILES; ++nt) {
            // B fragment: col = lo, elems 0..7 -> K = kb+0..7 ; elems 8..15 -> K = kb+16..23
            v16bf bfrag;
            const __bf16* Bp = &Wt[(nt * 16 + lo) * 36 + kb];
#pragma unroll
            for (int i = 0; i < 8; ++i) { bfrag[i] = Bp[i]; bfrag[i + 8] = Bp[i + 16]; }

            acc[nt] = __builtin_amdgcn_wmma_f32_16x16x32_bf16(
                false, afrag, false, bfrag, (short)0, acc[nt], false, false);
        }
    }

    if (!active) return;
#pragma unroll
    for (int nt = 0; nt < NTILES; ++nt) {
        float* Cp = C + (size_t)(m0 + hi * 8) * ldc + n0g + nt * 16 + lo;
#pragma unroll
        for (int r = 0; r < 8; ++r)
            Cp[(size_t)r * ldc] = acc[nt][r];
    }
}

// ---------------------------------------------------------------------------
// Degree / normalization
// ---------------------------------------------------------------------------
__global__ void deg_init_kernel(float* __restrict__ deg, int n) {
    int i = blockIdx.x * 256 + threadIdx.x;
    if (i < n) deg[i] = 1.0f;                 // + self loop
}
__global__ void deg_count_kernel(const int* __restrict__ dst, float* __restrict__ deg, int e) {
    int i = blockIdx.x * 256 + threadIdx.x;
    if (i < e) atomicAdd(&deg[dst[i]], 1.0f);
}
__global__ void deg_rsqrt_kernel(float* __restrict__ deg, int n) {
    int i = blockIdx.x * 256 + threadIdx.x;
    if (i < n) deg[i] = rsqrtf(deg[i]);
}

// ---------------------------------------------------------------------------
// Edge aggregation: one wave32 per edge; each lane moves 4 contiguous floats.
// agg[dst] += hw[src] * dinv[src]*dinv[dst]
// Edge endpoints forced scalar (wave-uniform) via readfirstlane.
// ---------------------------------------------------------------------------
__global__ __launch_bounds__(256)
void gcn_agg_kernel(const float* __restrict__ hw, const int* __restrict__ src,
                    const int* __restrict__ dst, const float* __restrict__ dinv,
                    float* __restrict__ agg, int nedge)
{
    int e = blockIdx.x * 8 + (threadIdx.x >> 5);
    if (e >= nedge) return;
    int lane = threadIdx.x & 31;
    int s = __builtin_amdgcn_readfirstlane(src[e]);
    int d = __builtin_amdgcn_readfirstlane(dst[e]);
    float c = dinv[s] * dinv[d];
    const float4 v = *(const float4*)(hw + (size_t)s * HDIM + lane * 4);
    float* ap = agg + (size_t)d * HDIM + lane * 4;
    atomicAdd(ap + 0, v.x * c);
    atomicAdd(ap + 1, v.y * c);
    atomicAdd(ap + 2, v.z * c);
    atomicAdd(ap + 3, v.w * c);
}

// self-loop + bias + relu (in place into agg)
__global__ void gcn_finish_kernel(float* __restrict__ agg, const float* __restrict__ hw,
                                  const float* __restrict__ dinv,
                                  const float* __restrict__ bias, int total)
{
    int i = blockIdx.x * 256 + threadIdx.x;
    if (i >= total) return;
    int row = i >> 7, col = i & 127;
    float di = dinv[row];
    float v = agg[i] + hw[i] * di * di + bias[col];
    agg[i] = v > 0.0f ? v : 0.0f;
}

// ---------------------------------------------------------------------------
// Elementwise helpers
// ---------------------------------------------------------------------------
__global__ void zero_kernel(float* __restrict__ p, int n) {
    int i = blockIdx.x * 256 + threadIdx.x;
    if (i < n) p[i] = 0.0f;
}
__global__ void bias_init_kernel(float* __restrict__ C, const float* __restrict__ bias,
                                 int total, int mask) {
    int i = blockIdx.x * 256 + threadIdx.x;
    if (i < total) C[i] = bias[i & mask];
}
__global__ void relu_kernel(float* __restrict__ p, int n) {
    int i = blockIdx.x * 256 + threadIdx.x;
    if (i < n) p[i] = fmaxf(p[i], 0.0f);
}

// Final head: out = sigmoid(z3 @ Wh4 + bh4), K = 64, N = 1
__global__ void head_final_kernel(const float* __restrict__ z3, const float* __restrict__ Wh4,
                                  const float* __restrict__ bh4, float* __restrict__ out, int nb)
{
    int b = blockIdx.x * 256 + threadIdx.x;
    if (b >= nb) return;
    const float* zp = z3 + (size_t)b * TDDIM;
    float s = bh4[0];
#pragma unroll
    for (int i = 0; i < TDDIM; ++i) s += zp[i] * Wh4[i];
    out[b] = 1.0f / (1.0f + __expf(-s));
}

// ---------------------------------------------------------------------------
extern "C" void kernel_launch(void* const* d_in, const int* in_sizes, int n_in,
                              void* d_out, int out_size, void* d_ws, size_t ws_size,
                              hipStream_t stream)
{
    (void)in_sizes; (void)n_in; (void)out_size; (void)ws_size;
    const float* x          = (const float*)d_in[0];
    const int*   edge_index = (const int*)d_in[1];
    const int*   src        = edge_index;
    const int*   dst        = edge_index + NEDGE;
    const int*   origin     = (const int*)d_in[2];
    const int*   destid     = (const int*)d_in[3];
    const int*   dayid      = (const int*)d_in[4];
    const int*   timeid     = (const int*)d_in[5];
    const int*   modeid     = (const int*)d_in[6];
    const float* W1 = (const float*)d_in[7];   const float* b1 = (const float*)d_in[8];
    const float* W2 = (const float*)d_in[9];   const float* b2 = (const float*)d_in[10];
    const float* W3 = (const float*)d_in[11];  const float* b3 = (const float*)d_in[12];
    const float* day_table  = (const float*)d_in[13];
    const float* time_table = (const float*)d_in[14];
    const float* mode_table = (const float*)d_in[15];
    const float* Wf  = (const float*)d_in[16]; const float* bf  = (const float*)d_in[17];
    const float* Wh1 = (const float*)d_in[18]; const float* bh1 = (const float*)d_in[19];
    const float* Wh2 = (const float*)d_in[20]; const float* bh2 = (const float*)d_in[21];
    const float* Wh3 = (const float*)d_in[22]; const float* bh3 = (const float*)d_in[23];
    const float* Wh4 = (const float*)d_in[24]; const float* bh4 = (const float*)d_in[25];
    float* out = (float*)d_out;

    // workspace layout (floats): dinv | buf0 | buf1 | buf2 | z1
    float* ws   = (float*)d_ws;
    float* dinv = ws;
    float* buf0 = ws + 100096;                        // hw scratch (all layers)
    float* buf1 = buf0 + (size_t)NNODES * HDIM;
    float* buf2 = buf1 + (size_t)NNODES * HDIM;
    float* z1   = buf2 + (size_t)NNODES * HDIM;       // [B, 256]
    float* temporal = buf0;                            // reuse after GCN
    float* z2   = buf2;                                // reuse after GCN
    float* z3   = buf2 + (size_t)NB * HDIM;

    const int totNH = NNODES * HDIM;

    // ---- symmetric normalization: dinv = rsqrt(deg + 1) -------------------
    deg_init_kernel<<<(NNODES + 255) / 256, 256, 0, stream>>>(dinv, NNODES);
    deg_count_kernel<<<(NEDGE + 255) / 256, 256, 0, stream>>>(dst, dinv, NEDGE);
    deg_rsqrt_kernel<<<(NNODES + 255) / 256, 256, 0, stream>>>(dinv, NNODES);

    dim3 gN((NNODES + 127) / 128, 1);
    dim3 gB1((NB + 127) / 128, 1);
    dim3 gB2((NB + 127) / 128, 2);

    // ---- GCN layer 1: x -> h1 (buf1) --------------------------------------
    gemm_bf16_wmma<8><<<gN, 256, 0, stream>>>(x, 128, nullptr, W1, 128, buf0, 128,
                                              NNODES, 128, 1);
    zero_kernel<<<(totNH + 255) / 256, 256, 0, stream>>>(buf1, totNH);
    gcn_agg_kernel<<<(NEDGE + 7) / 8, 256, 0, stream>>>(buf0, src, dst, dinv, buf1, NEDGE);
    gcn_finish_kernel<<<(totNH + 255) / 256, 256, 0, stream>>>(buf1, buf0, dinv, b1, totNH);

    // ---- GCN layer 2: h1 (buf1) -> h2 (buf2) ------------------------------
    gemm_bf16_wmma<8><<<gN, 256, 0, stream>>>(buf1, 128, nullptr, W2, 128, buf0, 128,
                                              NNODES, 128, 1);
    zero_kernel<<<(totNH + 255) / 256, 256, 0, stream>>>(buf2, totNH);
    gcn_agg_kernel<<<(NEDGE + 7) / 8, 256, 0, stream>>>(buf0, src, dst, dinv, buf2, NEDGE);
    gcn_finish_kernel<<<(totNH + 255) / 256, 256, 0, stream>>>(buf2, buf0, dinv, b2, totNH);

    // ---- GCN layer 3: h2 (buf2) -> h3 (buf1) ------------------------------
    gemm_bf16_wmma<8><<<gN, 256, 0, stream>>>(buf2, 128, nullptr, W3, 128, buf0, 128,
                                              NNODES, 128, 1);
    zero_kernel<<<(totNH + 255) / 256, 256, 0, stream>>>(buf1, totNH);
    gcn_agg_kernel<<<(NEDGE + 7) / 8, 256, 0, stream>>>(buf0, src, dst, dinv, buf1, NEDGE);
    gcn_finish_kernel<<<(totNH + 255) / 256, 256, 0, stream>>>(buf1, buf0, dinv, b3, totNH);
    float* hfin = buf1;

    // ---- temporal = relu([day_emb | time_emb] @ Wf + bf) (split-K gathers) -
    bias_init_kernel<<<(NB * 128 + 255) / 256, 256, 0, stream>>>(temporal, bf, NB * 128, 127);
    gemm_bf16_wmma<8><<<gB1, 256, 0, stream>>>(day_table, TDDIM, dayid, Wf, 128,
                                               temporal, 128, NB, TDDIM, 0);
    gemm_bf16_wmma<8><<<gB1, 256, 0, stream>>>(time_table, TDDIM, timeid, Wf + 64 * 128, 128,
                                               temporal, 128, NB, TDDIM, 0);
    relu_kernel<<<(NB * 128 + 255) / 256, 256, 0, stream>>>(temporal, NB * 128);

    // ---- z1 = relu(comb @ Wh1 + bh1) as 4 accumulate-GEMMs (no comb buffer)
    bias_init_kernel<<<(NB * 256 + 255) / 256, 256, 0, stream>>>(z1, bh1, NB * 256, 255);
    gemm_bf16_wmma<8><<<gB2, 256, 0, stream>>>(hfin, 128, origin, Wh1, 256,
                                               z1, 256, NB, 128, 0);
    gemm_bf16_wmma<8><<<gB2, 256, 0, stream>>>(hfin, 128, destid, Wh1 + 128 * 256, 256,
                                               z1, 256, NB, 128, 0);
    gemm_bf16_wmma<8><<<gB2, 256, 0, stream>>>(temporal, 128, nullptr, Wh1 + 256 * 256, 256,
                                               z1, 256, NB, 128, 0);
    gemm_bf16_wmma<8><<<gB2, 256, 0, stream>>>(mode_table, TDDIM, modeid, Wh1 + 384 * 256, 256,
                                               z1, 256, NB, TDDIM, 0);
    relu_kernel<<<(NB * 256 + 255) / 256, 256, 0, stream>>>(z1, NB * 256);

    // ---- z2 = relu(z1 @ Wh2 + bh2) ----------------------------------------
    bias_init_kernel<<<(NB * 128 + 255) / 256, 256, 0, stream>>>(z2, bh2, NB * 128, 127);
    gemm_bf16_wmma<8><<<gB1, 256, 0, stream>>>(z1, 256, nullptr, Wh2, 128,
                                               z2, 128, NB, 256, 0);
    relu_kernel<<<(NB * 128 + 255) / 256, 256, 0, stream>>>(z2, NB * 128);

    // ---- z3 = relu(z2 @ Wh3 + bh3) (N = 64 -> 4 column tiles) --------------
    bias_init_kernel<<<(NB * 64 + 255) / 256, 256, 0, stream>>>(z3, bh3, NB * 64, 63);
    gemm_bf16_wmma<4><<<gB1, 256, 0, stream>>>(z2, 128, nullptr, Wh3, 64,
                                               z3, 64, NB, 128, 0);
    relu_kernel<<<(NB * 64 + 255) / 256, 256, 0, stream>>>(z3, NB * 64);

    // ---- out = sigmoid(z3 @ Wh4 + bh4) ------------------------------------
    head_final_kernel<<<(NB + 255) / 256, 256, 0, stream>>>(z3, Wh4, bh4, out, NB);
}
